// ViTCatBeans_9990093931208
// MI455X (gfx1250) — matrix operand order repaired and checked by reading, not compiled, not run on hardware
//
#include <hip/hip_runtime.h>
#include <hip/hip_bf16.h>
#include <math.h>

// ---------------- static dims ----------------
#define BATCH 128
#define NPATCH 196
#define DDIM 512
#define CDD 256
#define SDD 256
#define VDD 256
#define FDD 2048
#define RROUTE 16
#define NHEAD1 8
#define NVERT 5
#define INDIM 768
#define M1 (BATCH * NPATCH)          // 25088 token rows (multiple of 128)
#define M2 (BATCH * RROUTE)          // 2048 fused rows  (multiple of 128)

typedef __bf16 bf16;
typedef bf16  bf16x16 __attribute__((ext_vector_type(16)));
typedef float f32x8   __attribute__((ext_vector_type(8)));

__device__ __forceinline__ float gelu_f(float x) {
    return 0.5f * x * (1.0f + erff(x * 0.70710678118654752440f));
}

// ================= generic bf16 WMMA GEMM =================
// C[M,N] = act( A[M,K](bf16,row-major) * W^T + bias[N] ),  W = Bt[N,K] row-major.
// Requires: K % 32 == 0, K % 8 == 0, M % 128 == 0.  N arbitrary (clamped loads,
// guarded stores).
// Block: 128 threads = 4 waves (2x2), block tile 128x128, wave tile 64x64,
// 4x4 WMMA tiles per wave -> 16 v_wmma per K-step, fragments amortized 4x.
#define SA 40   // LDS row stride in bf16 (80B: keeps 16B alignment, conflict-free
                // fragment gathers: row*20 banks distinct mod 64 for 16 lanes)

__global__ __launch_bounds__(128)
void gemm_bf16(const bf16* __restrict__ A, const bf16* __restrict__ Bt,
               const float* __restrict__ bias, float* __restrict__ C,
               int M, int N, int K, int act /*0 none,1 gelu*/)
{
    __shared__ __align__(16) bf16 lA[128 * SA];   // [m][k]  (128x32 tile)
    __shared__ __align__(16) bf16 lB[128 * SA];   // [n][k]  (128x32 tile) = W rows

    const int t     = threadIdx.x;
    const int wid   = t >> 5;
    const int lane  = t & 31;
    const int wm    = wid & 1;        // wave row (M)
    const int wn    = wid >> 1;       // wave col (N)
    const int m0    = blockIdx.y * 128;
    const int n0    = blockIdx.x * 128;
    const int l16   = lane & 15;
    const int lhalf = lane >> 4;

    f32x8 acc[4][4];
#pragma unroll
    for (int i = 0; i < 4; ++i)
#pragma unroll
        for (int j = 0; j < 4; ++j) acc[i][j] = (f32x8)0.0f;

    union FragU { unsigned u[8]; bf16x16 v; };

    // staging coords: 4 iterations x (row = lin>>2 in 0..127, ch = lin&3)
    const int srow = t >> 2;          // base row, +32 per iteration
    const int sch  = t & 3;           // 8-bf16 chunk within the 32-wide K slice

    for (int k0 = 0; k0 < K; k0 += 32) {
        // ---- stage A 128x32 (branch-free: M % 128 == 0) ----
#pragma unroll
        for (int it = 0; it < 4; ++it) {
            int row = srow + it * 32;
            uint4 val = *(const uint4*)(A + (size_t)(m0 + row) * K + k0 + sch * 8);
            *(uint4*)(&lA[row * SA + sch * 8]) = val;
        }
        // ---- stage B 128x32 straight from W[N,K] rows (clamped) ----
#pragma unroll
        for (int it = 0; it < 4; ++it) {
            int row = srow + it * 32;
            int gn  = n0 + row; gn = (gn < N) ? gn : (N - 1);
            uint4 val = *(const uint4*)(Bt + (size_t)gn * K + k0 + sch * 8);
            *(uint4*)(&lB[row * SA + sch * 8]) = val;
        }
        __syncthreads();

        // ---- fragments (ISA 16-bit A 16x32 / B 32x16 lane layouts) ----
        FragU a[4];
#pragma unroll
        for (int tm = 0; tm < 4; ++tm) {
            int row = wm * 64 + tm * 16 + l16;
#pragma unroll
            for (int i = 0; i < 8; ++i) {
                int ko = ((i < 4) ? 2 * i : 16 + 2 * (i - 4)) + lhalf * 8;
                a[tm].u[i] = *(const unsigned*)(&lA[row * SA + ko]);
            }
        }
#pragma unroll
        for (int tn = 0; tn < 4; ++tn) {
            FragU b;
            int col = wn * 64 + tn * 16 + l16;
#pragma unroll
            for (int j = 0; j < 8; ++j) {
                int ko = ((j < 4) ? 2 * j : 16 + 2 * (j - 4)) + lhalf * 8;
                b.u[j] = *(const unsigned*)(&lB[col * SA + ko]);
            }
#pragma unroll
            for (int tm = 0; tm < 4; ++tm)
                acc[tm][tn] = __builtin_amdgcn_wmma_f32_16x16x32_bf16(
                    false, a[tm].v, false, b.v,
                    (short)0, acc[tm][tn], false, false);
        }
        __syncthreads();
    }

    // ---- epilogue: bias + activation, f32 store ----
#pragma unroll
    for (int tn = 0; tn < 4; ++tn) {
        int col = n0 + wn * 64 + tn * 16 + l16;
        if (col >= N) continue;
        float bv = bias ? bias[col] : 0.0f;
#pragma unroll
        for (int tm = 0; tm < 4; ++tm) {
            int rbase = m0 + wm * 64 + tm * 16 + lhalf * 8;
#pragma unroll
            for (int r = 0; r < 8; ++r) {
                float v = acc[tm][tn][r] + bv;
                if (act == 1) v = gelu_f(v);
                C[(size_t)(rbase + r) * N + col] = v;
            }
        }
    }
}

// ================= conversion kernel (f32 -> bf16, flat) =================
__global__ void to_bf16_k(const float* __restrict__ in, bf16* __restrict__ out, size_t n) {
    size_t i = (size_t)blockIdx.x * 256 + threadIdx.x;
    if (i < n) out[i] = (bf16)in[i];
}

// ================= im2col for patch embed =================
__global__ void im2col_k(const float* __restrict__ x, bf16* __restrict__ out) {
    size_t idx = (size_t)blockIdx.x * 256 + threadIdx.x;
    if (idx >= (size_t)M1 * 768) return;
    int row = (int)(idx / 768), col = (int)(idx % 768);
    int b = row / NPATCH, n = row % NPATCH;
    int pr = n / 14, pc = n % 14;
    int c = col / 256, rem = col % 256;
    int iy = rem / 16, ix = rem % 16;
    out[idx] = (bf16)x[(((size_t)b * 3 + c) * 224 + pr * 16 + iy) * 224 + pc * 16 + ix];
}

__global__ void add_pos_k(float* __restrict__ tok, const float* __restrict__ pos) {
    size_t idx = (size_t)blockIdx.x * 256 + threadIdx.x;
    if (idx >= (size_t)M1 * DDIM) return;
    int row = (int)(idx / DDIM);
    int n = row % NPATCH, d = (int)(idx % DDIM);
    tok[idx] += pos[(size_t)n * DDIM + d];
}

// ================= attention 1: [B,196,512], 8 heads, dh=64 =================
__global__ __launch_bounds__(224)
void attn1_k(const float* __restrict__ qkv, float* __restrict__ o) {
    int bh = blockIdx.x;
    int b = bh >> 3, h = bh & 7;
    int i = threadIdx.x;
    if (i >= NPATCH) return;
    const size_t rowstride = 3 * DDIM;
    const float* qrow  = qkv + ((size_t)b * NPATCH + i) * rowstride + h * 64;
    const float* kbase = qkv + (size_t)b * NPATCH * rowstride + DDIM + h * 64;
    const float* vbase = qkv + (size_t)b * NPATCH * rowstride + 2 * DDIM + h * 64;
    float q[64], oa[64];
#pragma unroll
    for (int d = 0; d < 64; ++d) { q[d] = qrow[d]; oa[d] = 0.0f; }
    float mrun = -1e30f, lrun = 0.0f;
    for (int j = 0; j < NPATCH; ++j) {
        const float* kr = kbase + (size_t)j * rowstride;
        float s = 0.0f;
#pragma unroll
        for (int d = 0; d < 64; ++d) s += q[d] * kr[d];
        s *= 0.125f;                               // 64^-0.5
        float mn = fmaxf(mrun, s);
        float corr = __expf(mrun - mn);
        float p = __expf(s - mn);
        lrun = lrun * corr + p;
        const float* vr = vbase + (size_t)j * rowstride;
#pragma unroll
        for (int d = 0; d < 64; ++d) oa[d] = oa[d] * corr + p * vr[d];
        mrun = mn;
    }
    float inv = 1.0f / lrun;
    float* orow = o + ((size_t)b * NPATCH + i) * DDIM + h * 64;
#pragma unroll
    for (int d = 0; d < 64; ++d) orow[d] = oa[d] * inv;
}

// ================= attention 2: [B,16,768], 4 heads, dh=192 =================
__global__ __launch_bounds__(64)
void attn2_k(const float* __restrict__ qkv, float* __restrict__ o) {
    int b = blockIdx.x;
    int h = threadIdx.x >> 4;
    int i = threadIdx.x & 15;
    const size_t rowstride = 3 * INDIM;
    const float* qrow  = qkv + ((size_t)b * RROUTE + i) * rowstride + h * 192;
    const float* kbase = qkv + (size_t)b * RROUTE * rowstride + INDIM + h * 192;
    const float* vbase = qkv + (size_t)b * RROUTE * rowstride + 2 * INDIM + h * 192;
    float scale = rsqrtf(192.0f);
    float s[16];
    float mx = -1e30f;
#pragma unroll
    for (int j = 0; j < 16; ++j) {
        const float* kr = kbase + (size_t)j * rowstride;
        float a = 0.0f;
        for (int d = 0; d < 192; ++d) a += qrow[d] * kr[d];
        s[j] = a * scale;
        mx = fmaxf(mx, s[j]);
    }
    float sum = 0.0f;
#pragma unroll
    for (int j = 0; j < 16; ++j) { s[j] = __expf(s[j] - mx); sum += s[j]; }
    float inv = 1.0f / sum;
#pragma unroll
    for (int j = 0; j < 16; ++j) s[j] *= inv;
    float* orow = o + ((size_t)b * RROUTE + i) * INDIM + h * 192;
    for (int d = 0; d < 192; ++d) {
        float a = 0.0f;
#pragma unroll
        for (int j = 0; j < 16; ++j) a += s[j] * vbase[(size_t)j * rowstride + d];
        orow[d] = a;
    }
}

// ================= Cantor affinity [R,N] =================
__global__ __launch_bounds__(256)
void affinity_k(float* __restrict__ aff) {
    __shared__ float cx[NPATCH], cy[NPATCH];
    __shared__ float red[256];
    int t = threadIdx.x;
    if (t < NPATCH) {
        float i = (float)t;
        float w = floorf((sqrtf(8.0f * i + 1.0f) - 1.0f) * 0.5f);
        float tri = w * (w + 1.0f) * 0.5f;
        float y = i - tri;
        cx[t] = w - y; cy[t] = y;
    }
    __syncthreads();
    float mx = 0.0f;
    for (int p = t; p < NPATCH * NPATCH; p += 256) {
        int a = p / NPATCH, c = p % NPATCH;
        float dx = cx[a] - cx[c], dy = cy[a] - cy[c];
        mx = fmaxf(mx, sqrtf(dx * dx + dy * dy));
    }
    red[t] = mx; __syncthreads();
    for (int o = 128; o > 0; o >>= 1) {
        if (t < o) red[t] = fmaxf(red[t], red[t + o]);
        __syncthreads();
    }
    float dmax = red[0]; __syncthreads();
    for (int p = t; p < RROUTE * NPATCH; p += 256) {
        int r = p / NPATCH, n = p % NPATCH;
        float thr = 0.1f + 0.8f * (float)r / 15.0f;
        int cnt = 0;
        for (int m = 0; m < NPATCH; ++m) {
            float dx = cx[n] - cx[m], dy = cy[n] - cy[m];
            cnt += (sqrtf(dx * dx + dy * dy) / dmax < thr) ? 1 : 0;
        }
        aff[p] = (float)cnt / (float)NPATCH;
    }
}

// ================= LN over last dim 16 (one row per thread) =================
__global__ void ln16_k(float* __restrict__ X, const float* __restrict__ g,
                       const float* __restrict__ be, int rows) {
    int r = blockIdx.x * 256 + threadIdx.x;
    if (r >= rows) return;
    float* x = X + (size_t)r * 16;
    float m = 0.0f;
#pragma unroll
    for (int j = 0; j < 16; ++j) m += x[j];
    m *= (1.0f / 16.0f);
    float v = 0.0f;
#pragma unroll
    for (int j = 0; j < 16; ++j) { float d = x[j] - m; v += d * d; }
    v *= (1.0f / 16.0f);
    float inv = rsqrtf(v + 1e-5f);
#pragma unroll
    for (int j = 0; j < 16; ++j) x[j] = (x[j] - m) * inv * g[j] + be[j];
}

// ================= generic row LayerNorm (block per row) =================
__global__ __launch_bounds__(256)
void ln_rows_k(const float* __restrict__ in, float* __restrict__ out,
               const float* __restrict__ g, const float* __restrict__ be, int dim) {
    int row = blockIdx.x, t = threadIdx.x;
    const float* x = in + (size_t)row * dim;
    __shared__ float red[256];
    float s = 0.0f;
    for (int j = t; j < dim; j += 256) s += x[j];
    red[t] = s; __syncthreads();
    for (int o = 128; o > 0; o >>= 1) { if (t < o) red[t] += red[t + o]; __syncthreads(); }
    float mean = red[0] / (float)dim; __syncthreads();
    float v = 0.0f;
    for (int j = t; j < dim; j += 256) { float d = x[j] - mean; v += d * d; }
    red[t] = v; __syncthreads();
    for (int o = 128; o > 0; o >>= 1) { if (t < o) red[t] += red[t + o]; __syncthreads(); }
    float inv = rsqrtf(red[0] / (float)dim + 1e-5f);
    for (int j = t; j < dim; j += 256)
        out[(size_t)row * dim + j] = (x[j] - mean) * inv * g[j] + be[j];
}

// ================= routing softmax * affinity, renorm =================
__global__ __launch_bounds__(256)
void route_k(const float* __restrict__ logits /*[B,N,R], LN'ed*/,
             const float* __restrict__ aff /*[R,N]*/, float* __restrict__ rw /*[B,R,N]*/) {
    int b = blockIdx.x >> 4, r = blockIdx.x & 15;
    int t = threadIdx.x;
    __shared__ float red[256];
    float lv = (t < NPATCH) ? logits[((size_t)b * NPATCH + t) * RROUTE + r] : -1e30f;
    red[t] = lv; __syncthreads();
    for (int o = 128; o > 0; o >>= 1) { if (t < o) red[t] = fmaxf(red[t], red[t + o]); __syncthreads(); }
    float mx = red[0]; __syncthreads();
    float e = (t < NPATCH) ? __expf(lv - mx) : 0.0f;
    red[t] = e; __syncthreads();
    for (int o = 128; o > 0; o >>= 1) { if (t < o) red[t] += red[t + o]; __syncthreads(); }
    float sum = red[0]; __syncthreads();
    float wv = (t < NPATCH) ? (e / sum) * aff[r * NPATCH + t] : 0.0f;
    red[t] = wv; __syncthreads();
    for (int o = 128; o > 0; o >>= 1) { if (t < o) red[t] += red[t + o]; __syncthreads(); }
    float s2 = red[0] + 1e-8f;
    if (t < NPATCH) rw[((size_t)b * RROUTE + r) * NPATCH + t] = wv / s2;
}

// ================= routed reduction: out[b,r,:] = sum_n rw[b,r,n] X[b,n,:] ====
__global__ __launch_bounds__(256)
void routed_sum_k(const float* __restrict__ rw, const float* __restrict__ X,
                  float* __restrict__ out, int ldx) {
    int b = blockIdx.x >> 4, r = blockIdx.x & 15;
    int d = threadIdx.x;
    const float* w = rw + ((size_t)b * RROUTE + r) * NPATCH;
    const float* xb = X + (size_t)b * NPATCH * ldx + d;
    float s = 0.0f;
    for (int n = 0; n < NPATCH; ++n) s += w[n] * xb[(size_t)n * ldx];
    out[((size_t)b * RROUTE + r) * ldx + d] = s;
}

// ================= simplex barycentric projector =================
__global__ __launch_bounds__(64)
void simplex_k(const float* __restrict__ q, const float* __restrict__ verts,
               const float* __restrict__ w1, const float* __restrict__ b1,
               const float* __restrict__ w2, const float* __restrict__ b2,
               float* __restrict__ Q) {
    int row = blockIdx.x * 64 + threadIdx.x;
    if (row >= M2) return;
    const float* x = q + (size_t)row * SDD;
    float dots[NVERT] = {0, 0, 0, 0, 0};
    for (int d = 0; d < SDD; ++d) {
        float xv = x[d];
#pragma unroll
        for (int v = 0; v < NVERT; ++v) dots[v] += xv * verts[v * SDD + d];
    }
    float mx = dots[0];
#pragma unroll
    for (int v = 1; v < NVERT; ++v) mx = fmaxf(mx, dots[v]);
    float br[NVERT], s = 0.0f;
#pragma unroll
    for (int v = 0; v < NVERT; ++v) { br[v] = __expf(dots[v] - mx); s += br[v]; }
#pragma unroll
    for (int v = 0; v < NVERT; ++v) br[v] /= s;
    float h[2 * NVERT];
#pragma unroll
    for (int j = 0; j < 2 * NVERT; ++j) {
        float a = b1[j];
#pragma unroll
        for (int v = 0; v < NVERT; ++v) a += w1[j * NVERT + v] * br[v];
        h[j] = gelu_f(a);
    }
    float z[NVERT];
#pragma unroll
    for (int v = 0; v < NVERT; ++v) {
        float a = b2[v];
#pragma unroll
        for (int j = 0; j < 2 * NVERT; ++j) a += w2[v * 2 * NVERT + j] * h[j];
        z[v] = br[v] + a;
    }
    mx = z[0];
#pragma unroll
    for (int v = 1; v < NVERT; ++v) mx = fmaxf(mx, z[v]);
    s = 0.0f;
#pragma unroll
    for (int v = 0; v < NVERT; ++v) { z[v] = __expf(z[v] - mx); s += z[v]; }
#pragma unroll
    for (int v = 0; v < NVERT; ++v) z[v] /= s;
    for (int d = 0; d < SDD; ++d) {
        float a = 0.0f;
#pragma unroll
        for (int v = 0; v < NVERT; ++v) a += z[v] * verts[v * SDD + d];
        Q[(size_t)row * SDD + d] = a;
    }
}

// ================= V = V_route * (1 + sigmoid(gate_ln)) =================
__global__ void vgate_k(const float* __restrict__ vr, const float* __restrict__ gln,
                        float* __restrict__ V, size_t n) {
    size_t i = (size_t)blockIdx.x * 256 + threadIdx.x;
    if (i >= n) return;
    float gate = 1.0f / (1.0f + __expf(-gln[i]));
    V[i] = vr[i] * (1.0f + gate);
}

// ================= concat [K|Q|V] -> fused [M2,768] =================
__global__ void concat_k(const float* __restrict__ K, const float* __restrict__ Q,
                         const float* __restrict__ V, float* __restrict__ F) {
    size_t i = (size_t)blockIdx.x * 256 + threadIdx.x;
    if (i >= (size_t)M2 * INDIM) return;
    int row = (int)(i / INDIM), j = (int)(i % INDIM);
    float v;
    if (j < 256)      v = K[(size_t)row * 256 + j];
    else if (j < 512) v = Q[(size_t)row * 256 + (j - 256)];
    else              v = V[(size_t)row * 256 + (j - 512)];
    F[i] = v;
}

// ================= mean over R =================
__global__ void mean_r_k(const float* __restrict__ wide, float* __restrict__ Z) {
    int idx = blockIdx.x * 256 + threadIdx.x;
    if (idx >= BATCH * FDD) return;
    int b = idx / FDD, d = idx % FDD;
    float s = 0.0f;
#pragma unroll
    for (int r = 0; r < RROUTE; ++r) s += wide[((size_t)(b * RROUTE + r)) * FDD + d];
    Z[idx] = s * (1.0f / (float)RROUTE);
}

// ===================================================================
extern "C" void kernel_launch(void* const* d_in, const int* in_sizes, int n_in,
                              void* d_out, int out_size, void* d_ws, size_t ws_size,
                              hipStream_t stream) {
    const float* x        = (const float*)d_in[0];
    const float* conv_w   = (const float*)d_in[1];
    const float* conv_b   = (const float*)d_in[2];
    const float* pos      = (const float*)d_in[3];
    const float* m1_win   = (const float*)d_in[4];
    const float* m1_bin   = (const float*)d_in[5];
    const float* m1_wout  = (const float*)d_in[6];
    const float* m1_bout  = (const float*)d_in[7];
    const float* rm_w     = (const float*)d_in[8];
    const float* rm_b     = (const float*)d_in[9];
    const float* rm_g     = (const float*)d_in[10];
    const float* rm_be    = (const float*)d_in[11];
    const float* cantor_w = (const float*)d_in[12];
    const float* ts_w     = (const float*)d_in[13];
    const float* ts_b     = (const float*)d_in[14];
    const float* verts    = (const float*)d_in[15];
    const float* br1_w    = (const float*)d_in[16];
    const float* br1_b    = (const float*)d_in[17];
    const float* br2_w    = (const float*)d_in[18];
    const float* br2_b    = (const float*)d_in[19];
    const float* tv1_w    = (const float*)d_in[20];
    const float* tv1_b    = (const float*)d_in[21];
    const float* tv2_w    = (const float*)d_in[22];
    const float* tv2_b    = (const float*)d_in[23];
    const float* sg_w     = (const float*)d_in[24];
    const float* sg_b     = (const float*)d_in[25];
    const float* sg_g     = (const float*)d_in[26];
    const float* sg_be    = (const float*)d_in[27];
    const float* m2_win   = (const float*)d_in[28];
    const float* m2_bin   = (const float*)d_in[29];
    const float* m2_wout  = (const float*)d_in[30];
    const float* m2_bout  = (const float*)d_in[31];
    const float* fu1_w    = (const float*)d_in[32];
    const float* fu1_b    = (const float*)d_in[33];
    const float* fu2_w    = (const float*)d_in[34];
    const float* fu2_b    = (const float*)d_in[35];
    const float* pr_g     = (const float*)d_in[36];
    const float* pr_be    = (const float*)d_in[37];
    const float* pr_w     = (const float*)d_in[38];
    const float* pr_b     = (const float*)d_in[39];
    float* out = (float*)d_out;

    // ---- workspace bump allocator (256B aligned) ----
    size_t cur = 0;
    auto alloc = [&](size_t bytes) -> char* {
        char* p = (char*)d_ws + cur;
        cur += (bytes + 255) & ~(size_t)255;
        return p;
    };
    bf16* a_patch   = (bf16*)alloc((size_t)M1 * 768 * 2);
    bf16* w_patch   = (bf16*)alloc((size_t)DDIM * 768 * 2);         // [N=512,K=768]
    float* tok      = (float*)alloc((size_t)M1 * DDIM * 4);
    bf16* tok_bf    = (bf16*)alloc((size_t)M1 * DDIM * 2);
    bf16* w_qkv1    = (bf16*)alloc((size_t)3 * DDIM * DDIM * 2);    // [1536,512]
    float* qkv1     = (float*)alloc((size_t)M1 * 3 * DDIM * 4);
    float* o1       = (float*)alloc((size_t)M1 * DDIM * 4);
    bf16* o1_bf     = (bf16*)alloc((size_t)M1 * DDIM * 2);
    bf16* w_out1    = (bf16*)alloc((size_t)DDIM * DDIM * 2);
    float* ktok     = (float*)alloc((size_t)M1 * DDIM * 4);
    bf16* ktok_bf   = (bf16*)alloc((size_t)M1 * DDIM * 2);
    bf16* w_rm      = (bf16*)alloc((size_t)RROUTE * DDIM * 2);      // [16,512]
    float* logits   = (float*)alloc((size_t)M1 * RROUTE * 4);
    float* aff      = (float*)alloc((size_t)RROUTE * NPATCH * 4);
    float* routew   = (float*)alloc((size_t)M2 * NPATCH * 4);
    bf16* w_c       = (bf16*)alloc((size_t)CDD * DDIM * 2);         // [256,512]
    float* cantor   = (float*)alloc((size_t)M1 * CDD * 4);
    float* Kmat     = (float*)alloc((size_t)M2 * CDD * 4);
    bf16* K_bf      = (bf16*)alloc((size_t)M2 * CDD * 2);
    bf16* w_ts      = (bf16*)alloc((size_t)SDD * CDD * 2);          // [256,256]
    float* qsimp    = (float*)alloc((size_t)M2 * SDD * 4);
    float* Qmat     = (float*)alloc((size_t)M2 * SDD * 4);
    bf16* w_tv1     = (bf16*)alloc((size_t)2 * VDD * DDIM * 2);     // [512,512]
    float* hidtv    = (float*)alloc((size_t)M1 * 2 * VDD * 4);
    bf16* hidtv_bf  = (bf16*)alloc((size_t)M1 * 2 * VDD * 2);
    bf16* w_tv2     = (bf16*)alloc((size_t)VDD * 2 * VDD * 2);      // [256,512]
    float* Vraw     = (float*)alloc((size_t)M1 * VDD * 4);
    float* Vroute   = (float*)alloc((size_t)M2 * VDD * 4);
    bf16* Q_bf      = (bf16*)alloc((size_t)M2 * SDD * 2);
    bf16* w_sg      = (bf16*)alloc((size_t)VDD * SDD * 2);          // [256,256]
    float* sgraw    = (float*)alloc((size_t)M2 * VDD * 4);
    float* gateln   = (float*)alloc((size_t)M2 * VDD * 4);
    float* Vfin     = (float*)alloc((size_t)M2 * VDD * 4);
    float* fused    = (float*)alloc((size_t)M2 * INDIM * 4);
    bf16* fused_bf  = (bf16*)alloc((size_t)M2 * INDIM * 2);
    bf16* w_qkv2    = (bf16*)alloc((size_t)3 * INDIM * INDIM * 2);  // [2304,768]
    float* qkv2     = (float*)alloc((size_t)M2 * 3 * INDIM * 4);
    float* att2o    = (float*)alloc((size_t)M2 * INDIM * 4);
    bf16* att2o_bf  = (bf16*)alloc((size_t)M2 * INDIM * 2);
    bf16* w_out2    = (bf16*)alloc((size_t)INDIM * INDIM * 2);
    float* att2     = (float*)alloc((size_t)M2 * INDIM * 4);
    bf16* att2_bf   = (bf16*)alloc((size_t)M2 * INDIM * 2);
    bf16* w_fu1     = (bf16*)alloc((size_t)FDD * INDIM * 2);        // [2048,768]
    float* hid1     = (float*)alloc((size_t)M2 * FDD * 4);
    bf16* hid1_bf   = (bf16*)alloc((size_t)M2 * FDD * 2);
    bf16* w_fu2     = (bf16*)alloc((size_t)FDD * FDD * 2);          // [2048,2048]
    float* wide     = (float*)alloc((size_t)M2 * FDD * 4);
    float* Zm       = (float*)alloc((size_t)BATCH * FDD * 4);
    float* Zn       = (float*)alloc((size_t)BATCH * FDD * 4);
    bf16* Zn_bf     = (bf16*)alloc((size_t)BATCH * FDD * 2);
    bf16* w_pr      = (bf16*)alloc((size_t)FDD * FDD * 2);
    (void)ws_size; (void)in_sizes; (void)n_in; (void)out_size;

    auto cdiv = [](size_t a, size_t b) { return (int)((a + b - 1) / b); };
    auto gemm = [&](const bf16* A, const bf16* W, const float* bias, float* C,
                    int M, int N, int K, int act) {
        dim3 g((N + 127) / 128, M / 128);
        gemm_bf16<<<g, 128, 0, stream>>>(A, W, bias, C, M, N, K, act);
    };
    auto cvt = [&](const float* in, bf16* o, size_t n) {
        to_bf16_k<<<cdiv(n, 256), 256, 0, stream>>>(in, o, n);
    };

    // ---- weights: f32 (out,in) -> bf16, kept row-major (= transposed-B layout) ----
    cvt(conv_w,  w_patch, (size_t)DDIM * 768);
    cvt(m1_win,  w_qkv1,  (size_t)3 * DDIM * DDIM);
    cvt(m1_wout, w_out1,  (size_t)DDIM * DDIM);
    cvt(rm_w,    w_rm,    (size_t)RROUTE * DDIM);
    cvt(cantor_w,w_c,     (size_t)CDD * DDIM);
    cvt(ts_w,    w_ts,    (size_t)SDD * CDD);
    cvt(tv1_w,   w_tv1,   (size_t)2 * VDD * DDIM);
    cvt(tv2_w,   w_tv2,   (size_t)VDD * 2 * VDD);
    cvt(sg_w,    w_sg,    (size_t)VDD * SDD);
    cvt(m2_win,  w_qkv2,  (size_t)3 * INDIM * INDIM);
    cvt(m2_wout, w_out2,  (size_t)INDIM * INDIM);
    cvt(fu1_w,   w_fu1,   (size_t)FDD * INDIM);
    cvt(fu2_w,   w_fu2,   (size_t)FDD * FDD);
    cvt(pr_w,    w_pr,    (size_t)FDD * FDD);

    // ---- patch embed: im2col + GEMM + pos ----
    im2col_k<<<cdiv((size_t)M1 * 768, 256), 256, 0, stream>>>(x, a_patch);
    gemm(a_patch, w_patch, conv_b, tok, M1, DDIM, 768, 0);
    add_pos_k<<<cdiv((size_t)M1 * DDIM, 256), 256, 0, stream>>>(tok, pos);

    // ---- MHA1 -> k_tok ----
    cvt(tok, tok_bf, (size_t)M1 * DDIM);
    gemm(tok_bf, w_qkv1, m1_bin, qkv1, M1, 3 * DDIM, DDIM, 0);
    attn1_k<<<BATCH * NHEAD1, 224, 0, stream>>>(qkv1, o1);
    cvt(o1, o1_bf, (size_t)M1 * DDIM);
    gemm(o1_bf, w_out1, m1_bout, ktok, M1, DDIM, DDIM, 0);
    cvt(ktok, ktok_bf, (size_t)M1 * DDIM);

    // ---- routing ----
    gemm(ktok_bf, w_rm, rm_b, logits, M1, RROUTE, DDIM, 0);
    ln16_k<<<cdiv(M1, 256), 256, 0, stream>>>(logits, rm_g, rm_be, M1);
    affinity_k<<<1, 256, 0, stream>>>(aff);
    route_k<<<M2, 256, 0, stream>>>(logits, aff, routew);
    gemm(ktok_bf, w_c, nullptr, cantor, M1, CDD, DDIM, 0);
    routed_sum_k<<<M2, CDD, 0, stream>>>(routew, cantor, Kmat, CDD);

    // ---- simplex query ----
    cvt(Kmat, K_bf, (size_t)M2 * CDD);
    gemm(K_bf, w_ts, ts_b, qsimp, M2, SDD, CDD, 0);
    simplex_k<<<M2 / 64, 64, 0, stream>>>(qsimp, verts, br1_w, br1_b, br2_w, br2_b, Qmat);

    // ---- flow-matched value ----
    gemm(ktok_bf, w_tv1, tv1_b, hidtv, M1, 2 * VDD, DDIM, 1);
    cvt(hidtv, hidtv_bf, (size_t)M1 * 2 * VDD);
    gemm(hidtv_bf, w_tv2, tv2_b, Vraw, M1, VDD, 2 * VDD, 0);
    routed_sum_k<<<M2, VDD, 0, stream>>>(routew, Vraw, Vroute, VDD);
    cvt(Qmat, Q_bf, (size_t)M2 * SDD);
    gemm(Q_bf, w_sg, sg_b, sgraw, M2, VDD, SDD, 0);
    ln_rows_k<<<M2, 256, 0, stream>>>(sgraw, gateln, sg_g, sg_be, VDD);
    vgate_k<<<cdiv((size_t)M2 * VDD, 256), 256, 0, stream>>>(Vroute, gateln, Vfin,
                                                             (size_t)M2 * VDD);

    // ---- fusion core ----
    concat_k<<<cdiv((size_t)M2 * INDIM, 256), 256, 0, stream>>>(Kmat, Qmat, Vfin, fused);
    cvt(fused, fused_bf, (size_t)M2 * INDIM);
    gemm(fused_bf, w_qkv2, m2_bin, qkv2, M2, 3 * INDIM, INDIM, 0);
    attn2_k<<<BATCH, 64, 0, stream>>>(qkv2, att2o);
    cvt(att2o, att2o_bf, (size_t)M2 * INDIM);
    gemm(att2o_bf, w_out2, m2_bout, att2, M2, INDIM, INDIM, 0);
    cvt(att2, att2_bf, (size_t)M2 * INDIM);
    gemm(att2_bf, w_fu1, fu1_b, hid1, M2, FDD, INDIM, 1);
    cvt(hid1, hid1_bf, (size_t)M2 * FDD);
    gemm(hid1_bf, w_fu2, fu2_b, wide, M2, FDD, FDD, 0);

    // ---- head: mean over R, LN, proj + GELU ----
    mean_r_k<<<cdiv((size_t)BATCH * FDD, 256), 256, 0, stream>>>(wide, Zm);
    ln_rows_k<<<BATCH, 256, 0, stream>>>(Zm, Zn, pr_g, pr_be, FDD);
    cvt(Zn, Zn_bf, (size_t)BATCH * FDD);
    gemm(Zn_bf, w_pr, pr_b, out, BATCH, FDD, FDD, 1);
}
